// SwarmNet_83348135346678
// MI455X (gfx1250) — matrix operand order repaired and checked by reading, not compiled, not run on hardware
//
#include <hip/hip_runtime.h>

#define B_ 128
#define T_ 5
#define N_ 64
#define D_ 4
#define F1_ 32
#define F2_ 32
#define H_ 128
#define STEPS_ 8
#define TBUF_ 13  // T_ + STEPS_

// padded LDS row strides (bank-conflict avoidance: 64 banks x 4B = 256B period)
#define AH_LD 136   // bf16 elems: 272B row stride -> bank rotates 4/row
#define XB_LD 136   // bf16 elems (k_node X buffer)
#define YF_LD 132   // f32 elems  (k_node Y buffer)
#define XE_LD 40    // bf16 elems (encoder h buffer)

typedef __attribute__((ext_vector_type(16))) __bf16 v16bf;
typedef __attribute__((ext_vector_type(8)))  float  v8f;

__device__ __forceinline__ float relu_(float x) { return x > 0.f ? x : 0.f; }

// A-matrix 16x32 bf16 fragment from row-major LDS tile (ISA 7.12.2):
// lanes 0-15: row M=lane, K = 0..7 and 16..23 ; lanes 16-31: row M=lane-16, K = 8..15 and 24..31
__device__ __forceinline__ v16bf frag_a_lds(const __bf16* base, int ld, int lane) {
  int row = lane & 15;
  int kb  = (lane & 16) ? 8 : 0;
  const __bf16* p = base + row * ld + kb;
  v16bf f;
#pragma unroll
  for (int e = 0; e < 8; ++e) { f[e] = p[e]; f[e + 8] = p[16 + e]; }
  return f;
}

// B-matrix 32x16 bf16 fragment, loaded from W^T (N x K row-major):
// lane column N = lane&15 ; lanes 0-15 hold K=0..7,16..23, lanes 16-31 hold K=8..15,24..31
__device__ __forceinline__ v16bf frag_b_wT(const __bf16* baseT, int ld, int lane) {
  int col = lane & 15;
  int kb  = (lane & 16) ? 8 : 0;
  const __bf16* p = baseT + col * ld + kb;
  v16bf f;
#pragma unroll
  for (int e = 0; e < 8; ++e) { f[e] = p[e]; f[e + 8] = p[16 + e]; }
  return f;
}

__device__ __forceinline__ v8f wmma_bf16(v16bf a, v16bf b, v8f c) {
  return __builtin_amdgcn_wmma_f32_16x16x32_bf16(false, a, false, b, (short)0, c, false, false);
}

// gfx1250 async copy global->LDS (16B), tracked by ASYNCcnt (ISA 08 section 4)
__device__ __forceinline__ void async_load_b128(unsigned int lds_off, const void* gaddr) {
  asm volatile("global_load_async_to_lds_b128 %0, %1, off"
               :: "v"(lds_off), "v"((unsigned long long)(uintptr_t)gaddr) : "memory");
}
__device__ __forceinline__ void wait_asynccnt0() {
  asm volatile("s_wait_asynccnt 0x0" ::: "memory");
}

// ---------------------------------------------------------------- init ------
__global__ __launch_bounds__(256) void k_init(
    const float* __restrict__ ts, const float* __restrict__ ew1,
    const float* __restrict__ ew2, const float* __restrict__ nw1,
    const float* __restrict__ nw2, const float* __restrict__ dw1,
    const float* __restrict__ dw2, float* __restrict__ buf,
    __bf16* w1aT, __bf16* w1bT, __bf16* ew2T, __bf16* nw1T, __bf16* nw2T,
    __bf16* dw1bT, __bf16* dw2T) {
  int i = blockIdx.x * 256 + threadIdx.x;
  if (i < B_ * T_ * N_ * D_) {  // time_segs (B,T,N,D) -> buf (B,N,13,D) slots 0..4
    int d = i & 3; int r = i >> 2;
    int n = r % N_; r /= N_;
    int t = r % T_; int b = r / T_;
    buf[((b * N_ + n) * TBUF_ + t) * D_ + d] = ts[i];
    return;
  }
  i -= B_ * T_ * N_ * D_;
  if (i < H_ * F2_) { int k = i & 31, n = i >> 5; w1aT[n * F2_ + k] = (__bf16)ew1[k * H_ + n]; return; }
  i -= H_ * F2_;
  if (i < H_ * F2_) { int k = i & 31, n = i >> 5; w1bT[n * F2_ + k] = (__bf16)ew1[(F2_ + k) * H_ + n]; return; }
  i -= H_ * F2_;
  if (i < H_ * H_) { int k = i & 127, n = i >> 7; ew2T[n * H_ + k] = (__bf16)ew2[k * H_ + n]; return; }
  i -= H_ * H_;
  if (i < H_ * H_) { int k = i & 127, n = i >> 7; nw1T[n * H_ + k] = (__bf16)nw1[k * H_ + n]; return; }
  i -= H_ * H_;
  if (i < H_ * H_) { int k = i & 127, n = i >> 7; nw2T[n * H_ + k] = (__bf16)nw2[k * H_ + n]; return; }
  i -= H_ * H_;
  if (i < H_ * H_) { int k = i & 127, n = i >> 7; dw1bT[n * H_ + k] = (__bf16)dw1[(D_ + k) * H_ + n]; return; }
  i -= H_ * H_;
  if (i < H_ * H_) { int k = i & 127, n = i >> 7; dw2T[n * H_ + k] = (__bf16)dw2[k * H_ + n]; return; }
}

// --------------------------------------------------- encoder (convs + hu/hv)
__global__ __launch_bounds__(256) void k_encode(
    const float* __restrict__ buf, const float* __restrict__ cw1,
    const float* __restrict__ cb1, const float* __restrict__ cw2,
    const float* __restrict__ cb2, const __bf16* __restrict__ w1aT,
    const __bf16* __restrict__ w1bT, float* __restrict__ hu,
    float* __restrict__ hv, int step) {
  __shared__ float  xs[32][20];
  __shared__ float  c1[3 * 4 * 32];
  __shared__ float  c2[3 * 32 * 32];
  __shared__ float  y1[32][96];
  __shared__ __bf16 xb[32 * XE_LD];
  int tid = threadIdx.x;
  int g0  = blockIdx.x * 32;

  for (int i = tid; i < 384; i += 256)  c1[i] = cw1[i];
  for (int i = tid; i < 3072; i += 256) c2[i] = cw2[i];
  for (int i = tid; i < 32 * 20; i += 256) {
    int r = i / 20, q = i % 20;
    int g = g0 + r, b = g >> 6, n = g & 63;
    xs[r][q] = buf[((b * N_ + n) * TBUF_ + step) * D_ + q];  // 5 slots x 4 contiguous
  }
  __syncthreads();
  // conv1: (5,4) -> (3,32)
  for (int i = tid; i < 3072; i += 256) {
    int f1 = i & 31, tau = (i >> 5) % 3, r = i / 96;
    float s = cb1[f1];
#pragma unroll
    for (int j = 0; j < 3; ++j)
#pragma unroll
      for (int d = 0; d < 4; ++d)
        s += xs[r][(tau + j) * 4 + d] * c1[(j * 4 + d) * 32 + f1];
    y1[r][tau * 32 + f1] = relu_(s);
  }
  __syncthreads();
  // conv2: (3,32) -> (1,32)
  for (int i = tid; i < 1024; i += 256) {
    int f2 = i & 31, r = i >> 5;
    float s = cb2[f2];
    for (int q = 0; q < 96; ++q) s += y1[r][q] * c2[q * 32 + f2];
    xb[r * XE_LD + f2] = (__bf16)relu_(s);
  }
  __syncthreads();
  // hu = h @ ew1[:32], hv = h @ ew1[32:]  (32x32 @ 32x128)
  int lane = tid & 31, wave = tid >> 5;
  int mt  = wave >> 2;
  int ntb = (wave & 3) * 2;
  v16bf a = frag_a_lds(xb + (mt * 16) * XE_LD, XE_LD, lane);
#pragma unroll
  for (int u = 0; u < 2; ++u) {
    int nt = ntb + u;
    v8f accU, accV;
#pragma unroll
    for (int r = 0; r < 8; ++r) { accU[r] = 0.f; accV[r] = 0.f; }
    accU = wmma_bf16(a, frag_b_wT(w1aT + (nt * 16) * 32, 32, lane), accU);
    accV = wmma_bf16(a, frag_b_wT(w1bT + (nt * 16) * 32, 32, lane), accV);
    int col = nt * 16 + (lane & 15);
    int rb  = mt * 16 + ((lane & 16) ? 8 : 0);
#pragma unroll
    for (int r = 0; r < 8; ++r) {
      int g = g0 + rb + r;
      hu[g * H_ + col] = accU[r];
      hv[g * H_ + col] = accV[r];
    }
  }
}

// --------------------------------------- edge MLP + scatter (dominant GEMM) -
__global__ __launch_bounds__(256) void k_edge(
    const float* __restrict__ hu, const float* __restrict__ hv,
    const float* __restrict__ eb1, const float* __restrict__ eb2,
    const __bf16* __restrict__ ew2T, float* __restrict__ nmsg) {
  __shared__ float  hus[64 * 128];        // 32 KB: hu for all 64 sources of batch b
  __shared__ __bf16 Ah[64 * AH_LD];       // padded: relu(hu[s]+hv[t]+eb1) as bf16
  __shared__ float  tvec[128], b1s[128], b2s[128], colsum[128];
  int tid = threadIdx.x, bid = blockIdx.x;
  int b = bid >> 3, tbase = (bid & 7) * 8;
  int lane = tid & 31, wave = tid >> 5;
  int mt = wave >> 1, ntb = (wave & 1) * 4;

  // async global->LDS copy of the hu tile (8 x b128 per thread), ASYNCcnt-tracked
  {
    unsigned int lbase = (unsigned int)(uintptr_t)(&hus[0]);
    const float* g = hu + b * 8192;
#pragma unroll
    for (int j = 0; j < 8; ++j) {
      int i = (tid + j * 256) * 4;
      async_load_b128(lbase + i * 4, (const void*)(g + i));
    }
  }
  if (tid < 128) { b1s[tid] = eb1[tid]; b2s[tid] = eb2[tid]; }

  // hoist all 16 B fragments (4 N-tiles x 4 K-tiles) -- ew2T invariant over tt
  v16bf bw[16];
#pragma unroll
  for (int u = 0; u < 4; ++u)
#pragma unroll
    for (int kt = 0; kt < 4; ++kt)
      bw[u * 4 + kt] = frag_b_wT(ew2T + ((ntb + u) * 16) * 128 + kt * 32, 128, lane);

  wait_asynccnt0();
  __syncthreads();

  int rb = mt * 16 + ((lane & 16) ? 8 : 0);
  for (int tt = 0; tt < 8; ++tt) {
    int t = tbase + tt;
    if (tid < 128) { tvec[tid] = hv[(b * 64 + t) * 128 + tid]; colsum[tid] = 0.f; }
    if (tid == 0 && tt < 7) __builtin_prefetch(&hv[(b * 64 + t + 1) * 128], 0, 3);
    __syncthreads();
    for (int i = tid; i < 8192; i += 256) {
      int s = i >> 7, c = i & 127;
      Ah[s * AH_LD + c] = (__bf16)relu_(hus[i] + tvec[c] + b1s[c]);
    }
    __syncthreads();
    // 64x128 @ 128x128: per-wave 1 M-tile x 4 N-tiles, K unrolled x4
    v16bf a0 = frag_a_lds(Ah + (mt * 16) * AH_LD + 0,  AH_LD, lane);
    v16bf a1 = frag_a_lds(Ah + (mt * 16) * AH_LD + 32, AH_LD, lane);
    v16bf a2 = frag_a_lds(Ah + (mt * 16) * AH_LD + 64, AH_LD, lane);
    v16bf a3 = frag_a_lds(Ah + (mt * 16) * AH_LD + 96, AH_LD, lane);
#pragma unroll
    for (int u = 0; u < 4; ++u) {
      int nt = ntb + u;
      v8f acc;
#pragma unroll
      for (int r = 0; r < 8; ++r) acc[r] = 0.f;
      acc = wmma_bf16(a0, bw[u * 4 + 0], acc);
      acc = wmma_bf16(a1, bw[u * 4 + 1], acc);
      acc = wmma_bf16(a2, bw[u * 4 + 2], acc);
      acc = wmma_bf16(a3, bw[u * 4 + 3], acc);
      int col = nt * 16 + (lane & 15);
      float bias = b2s[col];
      float s = 0.f;
#pragma unroll
      for (int r = 0; r < 8; ++r) {
        float v = relu_(acc[r] + bias);
        if (rb + r != t) s += v;  // mask self-edge (adjacency = ones - eye)
      }
      atomicAdd(&colsum[col], s);  // ds_add_f32 scatter within block
    }
    __syncthreads();
    if (tid < 128) nmsg[(b * 64 + t) * 128 + tid] = colsum[tid];
    __syncthreads();
  }
}

// ------------------------------- node MLP + decoder + head + window update -
__global__ __launch_bounds__(256) void k_node(
    const float* __restrict__ nmsg, const __bf16* __restrict__ nw1T,
    const __bf16* __restrict__ nw2T, const __bf16* __restrict__ dw1bT,
    const __bf16* __restrict__ dw2T, const float* __restrict__ dw1,
    const float* __restrict__ ow, const float* __restrict__ nb1,
    const float* __restrict__ nb2, const float* __restrict__ db1,
    const float* __restrict__ db2, const float* __restrict__ ob,
    float* __restrict__ buf, float* __restrict__ out, int step) {
  __shared__ __bf16 Xb[32 * XB_LD];
  __shared__ float  Yf[32 * YF_LD];
  __shared__ float  prevs[32 * 4];
  __shared__ float  dw1a[4 * 128];  // prev-part of decoder W1 (K=4), f32
  __shared__ float  ows[128 * 4];
  int tid = threadIdx.x;
  int g0  = blockIdx.x * 32;
  int lane = tid & 31, wave = tid >> 5;
  int mt = wave >> 2, ntb = (wave & 3) * 2;

  for (int i = tid; i < 512; i += 256) { dw1a[i] = dw1[i]; ows[i] = ow[i]; }
  if (tid < 128) {
    int r = tid >> 2, d = tid & 3;
    int g = g0 + r, b = g >> 6, n = g & 63;
    prevs[tid] = buf[((b * N_ + n) * TBUF_ + step + 4) * D_ + d];
  }
  for (int i = tid; i < 4096; i += 256) {
    int r = i >> 7, c = i & 127;
    Xb[r * XB_LD + c] = (__bf16)nmsg[g0 * 128 + i];
  }
  __syncthreads();

  auto gemm_layer = [&](const __bf16* WT, const float* bias, bool addPrev) {
    const __bf16* ab = Xb + (mt * 16) * XB_LD;
    v16bf a0 = frag_a_lds(ab + 0,  XB_LD, lane);
    v16bf a1 = frag_a_lds(ab + 32, XB_LD, lane);
    v16bf a2 = frag_a_lds(ab + 64, XB_LD, lane);
    v16bf a3 = frag_a_lds(ab + 96, XB_LD, lane);
#pragma unroll
    for (int u = 0; u < 2; ++u) {
      int nt = ntb + u;
      v8f acc;
#pragma unroll
      for (int r = 0; r < 8; ++r) acc[r] = 0.f;
      const __bf16* wp = WT + (nt * 16) * 128;
      acc = wmma_bf16(a0, frag_b_wT(wp + 0,  128, lane), acc);
      acc = wmma_bf16(a1, frag_b_wT(wp + 32, 128, lane), acc);
      acc = wmma_bf16(a2, frag_b_wT(wp + 64, 128, lane), acc);
      acc = wmma_bf16(a3, frag_b_wT(wp + 96, 128, lane), acc);
      int col = nt * 16 + (lane & 15);
      int rb  = mt * 16 + ((lane & 16) ? 8 : 0);
      float bv = bias[col];
#pragma unroll
      for (int r = 0; r < 8; ++r) {
        int row = rb + r;
        float v = acc[r] + bv;
        if (addPrev) {
#pragma unroll
          for (int d = 0; d < 4; ++d) v += prevs[row * 4 + d] * dw1a[d * 128 + col];
        }
        Yf[row * YF_LD + col] = relu_(v);
      }
    }
  };
  auto y_to_x = [&]() {
    __syncthreads();
    for (int i = tid; i < 4096; i += 256) {
      int r = i >> 7, c = i & 127;
      Xb[r * XB_LD + c] = (__bf16)Yf[r * YF_LD + c];
    }
    __syncthreads();
  };

  gemm_layer(nw1T,  nb1, false); y_to_x();   // node MLP layer 1
  gemm_layer(nw2T,  nb2, false); y_to_x();   // node MLP layer 2
  gemm_layer(dw1bT, db1, true);  y_to_x();   // decoder layer 1 (+ prev @ dw1[:4])
  gemm_layer(dw2T,  db2, false);             // decoder layer 2
  __syncthreads();
  // head: nxt = d2 @ ow + ob + prev ; write window slot and output
  if (tid < 128) {
    int r = tid >> 2, d = tid & 3;
    float s = ob[d] + prevs[tid];
    const float* yr = &Yf[r * YF_LD];
    for (int c = 0; c < 128; ++c) s += yr[c] * ows[c * 4 + d];
    int g = g0 + r, b = g >> 6, n = g & 63;
    buf[((b * N_ + n) * TBUF_ + step + 5) * D_ + d] = s;
    out[((b * STEPS_ + step) * N_ + n) * D_ + d]    = s;
  }
}

// ----------------------------------------------------------------- launch --
extern "C" void kernel_launch(void* const* d_in, const int* in_sizes, int n_in,
                              void* d_out, int out_size, void* d_ws, size_t ws_size,
                              hipStream_t stream) {
  const float* ts  = (const float*)d_in[0];
  const float* cw1 = (const float*)d_in[1];
  const float* cb1 = (const float*)d_in[2];
  const float* cw2 = (const float*)d_in[3];
  const float* cb2 = (const float*)d_in[4];
  const float* ew1 = (const float*)d_in[5];
  const float* eb1 = (const float*)d_in[6];
  const float* ew2 = (const float*)d_in[7];
  const float* eb2 = (const float*)d_in[8];
  const float* nw1 = (const float*)d_in[9];
  const float* nb1 = (const float*)d_in[10];
  const float* nw2 = (const float*)d_in[11];
  const float* nb2 = (const float*)d_in[12];
  const float* dw1 = (const float*)d_in[13];
  const float* db1 = (const float*)d_in[14];
  const float* dw2 = (const float*)d_in[15];
  const float* db2 = (const float*)d_in[16];
  const float* ow  = (const float*)d_in[17];
  const float* ob  = (const float*)d_in[18];
  float* out = (float*)d_out;

  char* p = (char*)d_ws;
  float* buf  = (float*)p; p += sizeof(float) * B_ * N_ * TBUF_ * D_;
  float* hu   = (float*)p; p += sizeof(float) * B_ * N_ * H_;
  float* hv   = (float*)p; p += sizeof(float) * B_ * N_ * H_;
  float* nmsg = (float*)p; p += sizeof(float) * B_ * N_ * H_;
  __bf16* w1aT  = (__bf16*)p; p += sizeof(__bf16) * H_ * F2_;
  __bf16* w1bT  = (__bf16*)p; p += sizeof(__bf16) * H_ * F2_;
  __bf16* ew2T  = (__bf16*)p; p += sizeof(__bf16) * H_ * H_;
  __bf16* nw1T  = (__bf16*)p; p += sizeof(__bf16) * H_ * H_;
  __bf16* nw2T  = (__bf16*)p; p += sizeof(__bf16) * H_ * H_;
  __bf16* dw1bT = (__bf16*)p; p += sizeof(__bf16) * H_ * H_;
  __bf16* dw2T  = (__bf16*)p; p += sizeof(__bf16) * H_ * H_;

  // init work items: 163840 copy + 2*4096 + 5*16384 transposed bf16 = 253952
  k_init<<<992, 256, 0, stream>>>(ts, ew1, ew2, nw1, nw2, dw1, dw2, buf, w1aT,
                                  w1bT, ew2T, nw1T, nw2T, dw1bT, dw2T);
  for (int s = 0; s < STEPS_; ++s) {
    k_encode<<<(B_ * N_) / 32, 256, 0, stream>>>(buf, cw1, cb1, cw2, cb2, w1aT,
                                                 w1bT, hu, hv, s);
    k_edge<<<B_ * (N_ / 8), 256, 0, stream>>>(hu, hv, eb1, eb2, ew2T, nmsg);
    k_node<<<(B_ * N_) / 32, 256, 0, stream>>>(nmsg, nw1T, nw2T, dw1bT, dw2T,
                                               dw1, ow, nb1, nb2, db1, db2, ob,
                                               buf, out, s);
  }
}
